// SpatialEncoder_37701222924998
// MI455X (gfx1250) — matrix operand order repaired and checked by reading, not compiled, not run on hardware
//
#include <hip/hip_runtime.h>
#include <hip/hip_bf16.h>

// MI455X / gfx1250, wave32. WMMA f16->f32 for all GEMM-shaped work.

typedef __attribute__((ext_vector_type(16))) _Float16 v16h;
typedef __attribute__((ext_vector_type(8)))  _Float16 v8h;
typedef __attribute__((ext_vector_type(8)))  float    v8f;

#define B_ 1024
#define N_ 69
#define D_ 128
#define NEGV (-9000000.0f)

// Native gfx1250 transcendental ops: v_log_f32 (log2), v_exp_f32 (exp2).
#define FAST_LOG2(x) __builtin_amdgcn_logf(x)
#define FAST_EXP2(x) __builtin_amdgcn_exp2f(x)

// ---- LDS layout (dynamic shared, 134144 bytes total; gfx1250 has 320KB/WGP) ----
#define OFF_HSEM   0        // 80 x 128 f16  (20480)
#define OFF_HSEMT  20480    // 128 x 96 f16  (24576)  zero-padded cols 69..95
#define OFF_HSTR   45056    // 80 x 128 f16  (20480)
#define OFF_MSK1   65536    // 80 x 80  f32  (25600)
#define OFF_MSK2   91136    // 80 x 80  f32  (25600)
#define OFF_WBUF   116736   // 80 x 96  f16  (15360)  zero-padded
#define OFF_W4S    132096   // 4 x 128  f16  (1024)
#define OFF_W4T    133120   // 4 x 128  f16  (1024)
#define SMEM_BYTES 134144

__device__ inline v8f wmma16(v16h a, v16h b, v8f c) {
  // v_wmma_f32_16x16x32_f16 : D = A(16x32) x B(32x16) + C
  return __builtin_amdgcn_wmma_f32_16x16x32_f16(false, a, false, b, (short)0, c,
                                                false, false);
}

// A-matrix 16x32 f16 layout: lane L (m = L&15) holds K = (L>>4)*8 + {0..7} and
// K = 16 + (L>>4)*8 + {0..7}  -> two contiguous 8-half chunks.
__device__ inline v16h load_a_l(const _Float16* h, int ld, int row0, int k0, int lane) {
  int m = lane & 15, ks = (lane >> 4) * 8;
  const _Float16* p = h + (row0 + m) * ld + k0 + ks;
  v8h lo = *reinterpret_cast<const v8h*>(p);
  v8h hi = *reinterpret_cast<const v8h*>(p + 16);
  v16h r;
  for (int i = 0; i < 8; ++i) { r[i] = lo[i]; r[8 + i] = hi[i]; }
  return r;
}

// B-matrix 32x16 f16 layout: lane L (n = L&15) holds K = (L>>4)*8 + {0..7} and
// K = 16 + (L>>4)*8 + {0..7} of column n -> two contiguous 8-half chunks.
__device__ inline v16h load_b_l(const _Float16* h, int ld, int row0, int k0, int lane) {
  int n = lane & 15, ks = (lane >> 4) * 8;
  const _Float16* p = h + (row0 + n) * ld + k0 + ks;
  v8h lo = *reinterpret_cast<const v8h*>(p);
  v8h hi = *reinterpret_cast<const v8h*>(p + 16);
  v16h r;
  for (int i = 0; i < 8; ++i) { r[i] = lo[i]; r[8 + i] = hi[i]; }
  return r;
}

// A-layout fragment of a single 128-wide row vector (value independent of m).
__device__ inline v16h load_row_frag_l(const _Float16* wrow, int k0, int lane) {
  int ks = (lane >> 4) * 8;
  const _Float16* p = wrow + k0 + ks;
  v8h lo = *reinterpret_cast<const v8h*>(p);
  v8h hi = *reinterpret_cast<const v8h*>(p + 16);
  v16h r;
  for (int i = 0; i < 8; ++i) { r[i] = lo[i]; r[8 + i] = hi[i]; }
  return r;
}

// Global f32 -> f16 A fragment with row bound (zero-fill padding rows).
__device__ inline v16h load_a_g(const float* g, int ld, int row0, int k0, int lane,
                                int rowlim) {
  int m = lane & 15, ks = (lane >> 4) * 8;
  int row = row0 + m;
  v16h r;
  if (row < rowlim) {
    const float* p = g + (size_t)row * ld + k0 + ks;
    for (int i = 0; i < 8; ++i) {
      r[i] = (_Float16)p[i];
      r[8 + i] = (_Float16)p[16 + i];
    }
  } else {
    for (int i = 0; i < 16; ++i) r[i] = (_Float16)0.f;
  }
  return r;
}

// Global f32 -> f16 B fragment.
__device__ inline v16h load_b_g(const float* g, int ld, int row0, int k0, int lane) {
  int n = lane & 15, ks = (lane >> 4) * 8;
  const float* p = g + (size_t)(row0 + n) * ld + k0 + ks;
  v16h r;
  for (int i = 0; i < 8; ++i) { r[i] = (_Float16)p[i]; r[8 + i] = (_Float16)p[16 + i]; }
  return r;
}

__device__ inline float wave_sum(float v) {
  for (int m = 16; m > 0; m >>= 1) v += __shfl_xor(v, m, 32);
  return v;
}
__device__ inline float wave_max(float v) {
  for (int m = 16; m > 0; m >>= 1) v = fmaxf(v, __shfl_xor(v, m, 32));
  return v;
}
__device__ inline float pfn(float x, float tau, float invam1) {
  float t = x - tau;
  return (t > 0.f) ? FAST_EXP2(invam1 * FAST_LOG2(t)) : 0.f;
}

// ---------------------------------------------------------------------------
// Kernel 1: per-batch alphas.  Uses linearity of avg_pool for the structure
// branch so no big scratch GEMM is needed: avg_pool(X@W^T+b)+last = (m+l)@W^T+2b.
// ---------------------------------------------------------------------------
__global__ void __launch_bounds__(256) alpha_kernel(
    const float* __restrict__ in1, const float* __restrict__ in2,
    const float* __restrict__ SE, const float* __restrict__ mask,
    const int* __restrict__ iidx, const float* __restrict__ W_lin,
    const float* __restrict__ b_lin, const float* __restrict__ w1,
    const float* __restrict__ b1, const float* __restrict__ w2,
    const float* __restrict__ b2, const float* __restrict__ wst,
    const float* __restrict__ bst, float* __restrict__ alphas) {
  __shared__ float red[128];
  __shared__ float uvec[128];
  __shared__ float sdev;
  const int b = blockIdx.x, tid = threadIdx.x;
  const size_t base = (size_t)b * N_ * D_;
  const int last = iidx[b * N_ + (N_ - 1)];

  if (tid == 0) {
    float s = 0.f;
    for (int i = 0; i < N_; ++i) s += mask[b * N_ + i];
    sdev = s;
  }
  __syncthreads();
  const float dev = sdev;

  // ---- branch 1: input1 / w_sema1 ----
  if (tid < 128) {
    float s = 0.f;
    for (int i = 0; i < N_; ++i) s += in1[base + (size_t)i * D_ + tid] * mask[b * N_ + i];
    float t = s / dev + in1[base + (size_t)last * D_ + tid];
    red[tid] = t * w1[tid];
  }
  __syncthreads();
  for (int s = 64; s > 0; s >>= 1) {
    if (tid < s) red[tid] += red[tid + s];
    __syncthreads();
  }
  if (tid == 0) {
    float z = red[0] + b1[0];
    float r = 1.f / (1.f + expf(-z)) + 1.f;
    if (r == 1.f) r = 1.0001f;
    alphas[b * 3 + 0] = r;
  }
  __syncthreads();

  // ---- branch 2: input2 / w_sema2 ----
  if (tid < 128) {
    float s = 0.f;
    for (int i = 0; i < N_; ++i) s += in2[base + (size_t)i * D_ + tid] * mask[b * N_ + i];
    float t = s / dev + in2[base + (size_t)last * D_ + tid];
    red[tid] = t * w2[tid];
  }
  __syncthreads();
  for (int s = 64; s > 0; s >>= 1) {
    if (tid < s) red[tid] += red[tid + s];
    __syncthreads();
  }
  if (tid == 0) {
    float z = red[0] + b2[0];
    float r = 1.f / (1.f + expf(-z)) + 1.f;
    if (r == 1.f) r = 1.0001f;
    alphas[b * 3 + 1] = r;
  }
  __syncthreads();

  // ---- structure branch: u = masked-mean(SE) + SE[last]; target = u@W^T + 2b ----
  if (tid < 128) {
    float s = 0.f;
    for (int i = 0; i < N_; ++i) s += SE[base + (size_t)i * D_ + tid] * mask[b * N_ + i];
    uvec[tid] = s / dev + SE[base + (size_t)last * D_ + tid];
  }
  __syncthreads();
  if (tid < 128) {
    float t = 0.f;
    const float* wr = W_lin + (size_t)tid * D_;
    for (int k = 0; k < 128; ++k) t += uvec[k] * wr[k];
    t += 2.f * b_lin[tid];
    red[tid] = t * wst[tid];
  }
  __syncthreads();
  for (int s = 64; s > 0; s >>= 1) {
    if (tid < s) red[tid] += red[tid + s];
    __syncthreads();
  }
  if (tid == 0) {
    float z = red[0] + bst[0];
    float r = 1.f / (1.f + expf(-z)) + 1.f;
    if (r == 1.f) r = 1.0001f;
    alphas[b * 3 + 2] = r;
  }
}

// ---------------------------------------------------------------------------
// Kernel 2: one workgroup per (batch, branch). 256 threads = 8 wave32.
// ---------------------------------------------------------------------------
__global__ void __launch_bounds__(256) spatial_main_kernel(
    const float* __restrict__ in1, const float* __restrict__ in2,
    const float* __restrict__ SE, const int* __restrict__ adjS,
    const int* __restrict__ adjND, const float* __restrict__ W_lin,
    const float* __restrict__ b_lin, const float* __restrict__ A1,
    const float* __restrict__ S1, const float* __restrict__ A2,
    const float* __restrict__ S2, const float* __restrict__ alphas,
    float* __restrict__ out) {
  extern __shared__ char smem[];
  _Float16* hsem  = (_Float16*)(smem + OFF_HSEM);   // 80 x 128
  _Float16* hsemT = (_Float16*)(smem + OFF_HSEMT);  // 128 x 96
  _Float16* hstr  = (_Float16*)(smem + OFF_HSTR);   // 80 x 128
  float*    msk1  = (float*)(smem + OFF_MSK1);      // 80 x 80
  float*    msk2  = (float*)(smem + OFF_MSK2);      // 80 x 80
  _Float16* wbuf  = (_Float16*)(smem + OFF_WBUF);   // 80 x 96
  _Float16* w4s   = (_Float16*)(smem + OFF_W4S);    // 4 x 128
  _Float16* w4t   = (_Float16*)(smem + OFF_W4T);    // 4 x 128

  const int blk = blockIdx.x;
  const int b = blk >> 1, o = blk & 1;
  const int tid = threadIdx.x, lane = tid & 31, wave = tid >> 5;

  const float* hsrc = o ? in2 : in1;
  const int* adj = o ? adjND : adjS;
  const float* Wsem = o ? A2 : A1;
  const float* Wstr = o ? S2 : S1;
  const float am1_sem = alphas[b * 3 + o] - 1.0f;
  const float am1_str = alphas[b * 3 + 2] - 1.0f;

  const float* hb = hsrc + (size_t)b * (N_ * D_);
  const float* seb = SE + (size_t)b * (N_ * D_);
  const int* adjb = adj + (size_t)b * (N_ * N_);

  // ---------- Phase A: cooperative staging (f32 -> f16), zero padding ----------
  for (int e = tid; e < 80 * 128; e += 256) {
    int i = e >> 7, d = e & 127;
    hsem[e] = (i < N_) ? (_Float16)hb[(size_t)i * D_ + d] : (_Float16)0.f;
  }
  for (int e = tid; e < 128 * 96; e += 256) {
    int d = e / 96, j = e - d * 96;
    hsemT[e] = (j < N_) ? (_Float16)hb[(size_t)j * D_ + d] : (_Float16)0.f;
  }
  for (int e = tid; e < 4 * 128; e += 256) {
    w4s[e] = (_Float16)Wsem[e];
    w4t[e] = (_Float16)Wstr[e];
  }
  for (int e = tid; e < 80 * 96; e += 256) wbuf[e] = (_Float16)0.f;

  // ---------- Phase B: hstr = SE_b @ W_lin^T + b_lin   (WMMA, 5x8 tiles) ----------
  for (int task = wave; task < 40; task += 8) {
    int it = task / 8, dt = task - it * 8;
    v8f acc = {};
    for (int ks = 0; ks < 4; ++ks) {
      int k0 = ks * 32;
      v16h a = load_a_g(seb, D_, it * 16, k0, lane, N_);
      v16h bf = load_b_g(W_lin, D_, dt * 16, k0, lane);  // B[k][n]=W_lin[n][k]
      acc = wmma16(a, bf, acc);
    }
    int n = lane & 15, rb = (lane < 16) ? 0 : 8;
    int dcol = dt * 16 + n;
    float bias = b_lin[dcol];
    for (int r = 0; r < 8; ++r) {
      int i = it * 16 + rb + r;
      hstr[i * D_ + dcol] = (_Float16)(acc[r] + bias);
    }
  }
  __syncthreads();

  // ---------- Phase C: e[k] = (h .* W4[k]) @ h^T, adj-select, leaky, mask ----------
  for (int task = wave; task < 50; task += 8) {
    int m = (task >= 25) ? 1 : 0;
    int tt = task - 25 * m;
    int it = tt / 5, jt = tt - it * 5;
    const _Float16* hX = m ? hstr : hsem;
    const _Float16* w4 = m ? w4t : w4s;
    float* mk = m ? msk2 : msk1;
    v8f a0 = {}, a1 = {}, a2 = {}, a3 = {};
    for (int ks = 0; ks < 4; ++ks) {
      int k0 = ks * 32;
      v16h a = load_a_l(hX, D_, it * 16, k0, lane);
      v16h bf = load_b_l(hX, D_, jt * 16, k0, lane);  // B[k=d][n=j] = h[j][d]
      v16h w0 = load_row_frag_l(w4 + 0 * D_, k0, lane);
      v16h w1 = load_row_frag_l(w4 + 1 * D_, k0, lane);
      v16h w2 = load_row_frag_l(w4 + 2 * D_, k0, lane);
      v16h w3 = load_row_frag_l(w4 + 3 * D_, k0, lane);
      a0 = wmma16(a * w0, bf, a0);  // v_pk_mul_f16 scale of A per k
      a1 = wmma16(a * w1, bf, a1);
      a2 = wmma16(a * w2, bf, a2);
      a3 = wmma16(a * w3, bf, a3);
    }
    int n = lane & 15, rb = (lane < 16) ? 0 : 8;
    int j = jt * 16 + n;
    for (int r = 0; r < 8; ++r) {
      int i = it * 16 + rb + r;
      float mval = NEGV;
      if (i < N_ && j < N_) {
        int av = adjb[i * N_ + j];
        if (av >= 1) {
          int id = av - 1;
          if (id > 3) id = 3;
          float e = (id == 0) ? a0[r] : (id == 1) ? a1[r] : (id == 2) ? a2[r] : a3[r];
          mval = (e >= 0.f) ? e : 0.2f * e;
        }
      }
      mk[i * 80 + j] = mval;
    }
  }
  __syncthreads();

  // ---------- Phase D: entmax bisection (50 iters), row per wave, in place ----------
  for (int t = wave; t < 2 * N_; t += 8) {
    int m = (t >= N_) ? 1 : 0;
    int i = t - N_ * m;
    float* row = (m ? msk2 : msk1) + i * 80;
    float am1 = m ? am1_str : am1_sem;
    float inv = 1.0f / am1;
    int j0 = lane, j1 = lane + 32, j2 = lane + 64;
    bool has2 = (j2 < 80);
    float x0 = row[j0] * am1;
    float x1 = row[j1] * am1;
    float x2 = has2 ? row[j2] * am1 : -3.0e38f;
    float mx = wave_max(fmaxf(x0, fmaxf(x1, x2)));
    float tau_lo = mx - 1.0f;
    float tau_hi = mx - FAST_EXP2(am1 * FAST_LOG2(1.0f / (float)N_));
    float dm = tau_hi - tau_lo;
    float tau = tau_lo;
    for (int s = 0; s < 50; ++s) {
      dm *= 0.5f;
      tau = tau_lo + dm;
      float f = pfn(x0, tau, inv) + pfn(x1, tau, inv) + pfn(x2, tau, inv);
      f = wave_sum(f) - 1.0f;
      if (f >= 0.f) tau_lo = tau;  // uniform across wave
    }
    float p0 = pfn(x0, tau, inv);
    float p1 = pfn(x1, tau, inv);
    float p2 = pfn(x2, tau, inv);
    float s = wave_sum(p0 + p1 + p2);
    float is = 1.0f / s;
    row[j0] = p0 * is;
    row[j1] = p1 * is;
    if (has2) row[j2] = p2 * is;
  }
  __syncthreads();

  // ---------- Phase E: w = p1*p2, row-normalize, f16 into zero-padded wbuf ----------
  for (int i = wave; i < N_; i += 8) {
    int j0 = lane, j1 = lane + 32, j2 = lane + 64;
    bool has2 = (j2 < 80);
    float w0 = msk1[i * 80 + j0] * msk2[i * 80 + j0];
    float w1v = msk1[i * 80 + j1] * msk2[i * 80 + j1];
    float w2v = has2 ? msk1[i * 80 + j2] * msk2[i * 80 + j2] : 0.f;
    float s = wave_sum(w0 + w1v + w2v);
    float is = 1.0f / (s + 1e-7f);
    wbuf[i * 96 + j0] = (_Float16)(w0 * is);
    wbuf[i * 96 + j1] = (_Float16)(w1v * is);
    if (has2) wbuf[i * 96 + j2] = (_Float16)(w2v * is);  // j>=69 entries are 0
  }
  __syncthreads();

  // ---------- Phase F: out = w @ h_sem   (K = 96 padded, WMMA) ----------
  for (int task = wave; task < 40; task += 8) {
    int it = task / 8, dt = task - it * 8;
    v8f acc = {};
    for (int ks = 0; ks < 3; ++ks) {
      int k0 = ks * 32;
      v16h a = load_a_l(wbuf, 96, it * 16, k0, lane);
      v16h bf = load_b_l(hsemT, 96, dt * 16, k0, lane);  // B[k=j][n=d]=h[j][d]
      acc = wmma16(a, bf, acc);
    }
    int n = lane & 15, rb = (lane < 16) ? 0 : 8;
    int dcol = dt * 16 + n;
    for (int r = 0; r < 8; ++r) {
      int i = it * 16 + rb + r;
      if (i < N_)
        out[((size_t)o * B_ * N_ + (size_t)b * N_ + i) * D_ + dcol] = acc[r];
    }
  }
}

extern "C" void kernel_launch(void* const* d_in, const int* in_sizes, int n_in,
                              void* d_out, int out_size, void* d_ws, size_t ws_size,
                              hipStream_t stream) {
  (void)in_sizes; (void)n_in; (void)out_size; (void)ws_size;
  const float* in1   = (const float*)d_in[0];
  const float* in2   = (const float*)d_in[1];
  const float* SE    = (const float*)d_in[2];
  const int*   adjS  = (const int*)d_in[3];
  const int*   adjND = (const int*)d_in[4];
  const float* mask  = (const float*)d_in[5];
  const int*   iidx  = (const int*)d_in[6];
  const float* W_lin = (const float*)d_in[8];
  const float* b_lin = (const float*)d_in[9];
  const float* w1    = (const float*)d_in[10];
  const float* b1    = (const float*)d_in[11];
  const float* w2    = (const float*)d_in[12];
  const float* b2    = (const float*)d_in[13];
  const float* wst   = (const float*)d_in[14];
  const float* bst   = (const float*)d_in[15];
  const float* A1    = (const float*)d_in[16];
  const float* S1    = (const float*)d_in[17];
  const float* A2    = (const float*)d_in[18];
  const float* S2    = (const float*)d_in[19];
  float* alphas = (float*)d_ws;  // 3*B floats of scratch
  float* out = (float*)d_out;

  alpha_kernel<<<B_, 256, 0, stream>>>(in1, in2, SE, mask, iidx, W_lin, b_lin,
                                       w1, b1, w2, b2, wst, bst, alphas);
  spatial_main_kernel<<<2 * B_, 256, SMEM_BYTES, stream>>>(
      in1, in2, SE, adjS, adjND, W_lin, b_lin, A1, S1, A2, S2, alphas, out);
}
